// GraphTripleConv_67181878444956
// MI455X (gfx1250) — matrix operand order repaired and checked by reading, not compiled, not run on hardware
//
#include <hip/hip_runtime.h>
#include <hip/hip_bf16.h>
#include <cstdint>
#include <cstddef>

// ---------------------------------------------------------------------------
// Problem constants (from the reference)
// ---------------------------------------------------------------------------
#define B_DIM   32
#define T_DIM   2048
#define O_DIM   18
#define DIN1_K  1500
#define H_DIM   512
#define DOUT    512
#define MT      (B_DIM * T_DIM)      /* 65536 rows for the big GEMMs */
#define OH      (O_DIM * H_DIM)      /* 9216 */
#define NCHUNK  8

typedef __attribute__((ext_vector_type(16))) __bf16 v16bf;
typedef __attribute__((ext_vector_type(8)))  float  v8f;

// ---------------------------------------------------------------------------
// bf16 helpers (RNE split: v = hi + lo, ~16 effective mantissa bits)
// ---------------------------------------------------------------------------
__device__ __forceinline__ unsigned short f2bf(float f) {
  unsigned int u = __float_as_uint(f);
  unsigned int r = (u + 0x7FFFu + ((u >> 16) & 1u)) >> 16;
  return (unsigned short)r;
}
__device__ __forceinline__ float bf2f(unsigned short h) {
  return __uint_as_float(((unsigned int)h) << 16);
}

union Frag {
  v16bf v;
  unsigned short u[16];
  uint4 q[2];
};

// CDNA5 async global->LDS copy (16 B per lane), tracked by ASYNCcnt.
__device__ __forceinline__ void async_cp16(unsigned ldsaddr, const void* gaddr) {
  asm volatile("global_load_async_to_lds_b128 %0, %1, off"
               :: "v"(ldsaddr), "v"(gaddr)
               : "memory");
}
__device__ __forceinline__ void wait_async0() {
  asm volatile("s_wait_asynccnt 0x0" ::: "memory");
}
// Flat LDS pointer -> 32-bit wave-relative LDS address (low 32 bits of flat).
__device__ __forceinline__ unsigned lds_off(const void* p) {
  return (unsigned)(uintptr_t)p;
}

// ---------------------------------------------------------------------------
// Weight prep: W (K x N, fp32 row-major) -> W^T hi/lo bf16 [N][Kpad],
// zero-filled for k >= K so the GEMM K-loop needs no bounds checks on B.
// ---------------------------------------------------------------------------
__global__ void split_weight_T(const float* __restrict__ W, int K, int N, int Kpad,
                               unsigned short* __restrict__ hi,
                               unsigned short* __restrict__ lo, int total) {
  int idx = blockIdx.x * blockDim.x + threadIdx.x;
  if (idx >= total) return;
  int n = idx / Kpad;
  int k = idx - n * Kpad;
  float v = (k < K) ? W[(size_t)k * N + n] : 0.0f;
  unsigned short h = f2bf(v);
  hi[idx] = h;
  lo[idx] = f2bf(v - bf2f(h));
}

// ---------------------------------------------------------------------------
// Split-bf16 WMMA GEMM, double-buffered, async B staging.
//   D = relu(A @ B + bias); A fp32 (M x K, lda), B pre-split/transposed bf16.
//   Block: 256 threads = 8 waves; block tile 64(M) x 256(N); K step 32.
//   Wave w: rows (w>>1)*16, cols (w&1)*128 -> 8 accumulators, 24 WMMA/K-step
//   (hi*hi + hi*lo + lo*hi per 16x16x32 tile -> ~fp32 accuracy).
// ---------------------------------------------------------------------------
__global__ void __launch_bounds__(256)
gemm_bf16x3_relu(const float* __restrict__ A, int lda, int K, int Kpad,
                 const unsigned short* __restrict__ BTh,
                 const unsigned short* __restrict__ BTl,
                 const float* __restrict__ bias,
                 float* __restrict__ dst0, float* __restrict__ dst1,
                 float* __restrict__ dst2,
                 int split1, int split2, int ld0, int ld1, int ld2) {
  // Row stride 40 halves (80 B): every 16-B chunk stays 16-B aligned and the
  // 16 rows of a fragment spread over all 64 LDS banks (20 dwords/row).
  __shared__ __align__(16) unsigned short Ah[2][64][40];
  __shared__ __align__(16) unsigned short Al[2][64][40];
  __shared__ __align__(16) unsigned short Bh[2][256][40];
  __shared__ __align__(16) unsigned short Bl[2][256][40];

  const int tid  = threadIdx.x;
  const int lane = tid & 31;
  const int wave = tid >> 5;
  const int mg   = wave >> 1;            // 0..3 -> wave row group (16 rows)
  const int ng   = (wave & 1) * 128;     // wave col half within 256
  const int m0   = blockIdx.y * 64;
  const int n0   = blockIdx.x * 256;

  v8f zero = {0.f, 0.f, 0.f, 0.f, 0.f, 0.f, 0.f, 0.f};
  v8f acc[8];
#pragma unroll
  for (int j = 0; j < 8; ++j) acc[j] = zero;

  // ---- staging maps ----
  const int arow  = tid >> 2;            // 0..63
  const int acol8 = (tid & 3) * 8;       // 0,8,16,24
  const size_t abase = (size_t)(m0 + arow) * (size_t)lda;

  // B: 4 16-B chunks per thread per matrix, per K-step (256 x 32 halves)
  auto stageB = [&](int nb, int k0) {
#pragma unroll
    for (int j = 0; j < 4; ++j) {
      int cc  = tid + j * 256;
      int row = cc >> 2;                 // 0..255
      int kc  = (cc & 3) * 8;
      size_t g = (size_t)(n0 + row) * (size_t)Kpad + (size_t)(k0 + kc);
      async_cp16(lds_off(&Bh[nb][row][kc]), BTh + g);
      async_cp16(lds_off(&Bl[nb][row][kc]), BTl + g);
    }
  };
  // A: issue loads (8 fp32 per thread) -- vectorized unless K-tail tile.
  auto loadA = [&](float va[8], int k0) {
    if (k0 + 32 <= K) {
      const float* ap = A + abase + k0 + acol8;
      float4 x = *(const float4*)ap;
      float4 y = *(const float4*)(ap + 4);
      va[0] = x.x; va[1] = x.y; va[2] = x.z; va[3] = x.w;
      va[4] = y.x; va[5] = y.y; va[6] = y.z; va[7] = y.w;
    } else {
#pragma unroll
      for (int i = 0; i < 8; ++i) {
        int k = k0 + acol8 + i;
        va[i] = (k < K) ? A[abase + k] : 0.0f;
      }
    }
  };
  // A: split to hi/lo bf16 and store to LDS (2 x 16 B per thread).
  auto storeA = [&](int ab, const float va[8]) {
    uint4 hq, lq;
    unsigned hu[4], lu[4];
#pragma unroll
    for (int i = 0; i < 4; ++i) {
      unsigned short h0 = f2bf(va[2 * i]);
      unsigned short h1 = f2bf(va[2 * i + 1]);
      hu[i] = (unsigned)h0 | ((unsigned)h1 << 16);
      lu[i] = (unsigned)f2bf(va[2 * i] - bf2f(h0)) |
              ((unsigned)f2bf(va[2 * i + 1] - bf2f(h1)) << 16);
    }
    hq.x = hu[0]; hq.y = hu[1]; hq.z = hu[2]; hq.w = hu[3];
    lq.x = lu[0]; lq.y = lu[1]; lq.z = lu[2]; lq.w = lu[3];
    *(uint4*)&Ah[ab][arow][acol8] = hq;
    *(uint4*)&Al[ab][arow][acol8] = lq;
  };

  // ---- prologue: stage tile 0 ----
  {
    float va[8];
    stageB(0, 0);
    loadA(va, 0);
    storeA(0, va);
  }
  wait_async0();
  __syncthreads();

  const int nk = Kpad / 32;
  for (int it = 0; it < nk; ++it) {
    const int buf = it & 1;
    const bool more = (it + 1) < nk;

    float va[8];
    if (more) {
      stageB(buf ^ 1, (it + 1) * 32);   // async: streams during compute
      loadA(va, (it + 1) * 32);         // issue now, consume after compute
    }

    // ---- compute tile `buf` ----
    {
      const int m   = mg * 16 + (lane & 15);
      const int kbA = (lane >> 4) * 8;   // A: lane<16 -> K{0..7,16..23}
      Frag ah, al;
      ah.q[0] = *(const uint4*)&Ah[buf][m][kbA];
      ah.q[1] = *(const uint4*)&Ah[buf][m][kbA + 16];
      al.q[0] = *(const uint4*)&Al[buf][m][kbA];
      al.q[1] = *(const uint4*)&Al[buf][m][kbA + 16];
      const int kbB = (lane >> 4) * 8;   // B: lane<16 -> K{0..7,16..23} at N=lane
#pragma unroll
      for (int j = 0; j < 8; ++j) {
        const int n = ng + j * 16 + (lane & 15);
        Frag bh, bl;
        bh.q[0] = *(const uint4*)&Bh[buf][n][kbB];
        bh.q[1] = *(const uint4*)&Bh[buf][n][kbB + 16];
        bl.q[0] = *(const uint4*)&Bl[buf][n][kbB];
        bl.q[1] = *(const uint4*)&Bl[buf][n][kbB + 16];
        acc[j] = __builtin_amdgcn_wmma_f32_16x16x32_bf16(
                     false, ah.v, false, bh.v, (short)0, acc[j], false, false);
        acc[j] = __builtin_amdgcn_wmma_f32_16x16x32_bf16(
                     false, ah.v, false, bl.v, (short)0, acc[j], false, false);
        acc[j] = __builtin_amdgcn_wmma_f32_16x16x32_bf16(
                     false, al.v, false, bh.v, (short)0, acc[j], false, false);
      }
    }

    if (more) storeA(buf ^ 1, va);      // convert + ds_store after compute
    wait_async0();
    __syncthreads();                    // also flushes DScnt before reuse
  }

  // ---- epilogue: bias + ReLU + column-routed store ----
  const int colLane = lane & 15;
  const int rowOff  = (lane >> 4) * 8;  // C/D: lanes>=16 hold M=8..15
#pragma unroll
  for (int j = 0; j < 8; ++j) {
    const int gcol = n0 + ng + j * 16 + colLane;
    const float bia = bias[gcol];
#pragma unroll
    for (int r = 0; r < 8; ++r) {
      const int grow = m0 + mg * 16 + rowOff + r;
      float v = acc[j][r] + bia;
      v = v > 0.f ? v : 0.f;
      if (gcol < split1)
        dst0[(size_t)grow * ld0 + gcol] = v;
      else if (gcol < split2)
        dst1[(size_t)grow * ld1 + (gcol - split1)] = v;
      else
        dst2[(size_t)grow * ld2 + (gcol - split2)] = v;
    }
  }
}

// ---------------------------------------------------------------------------
// Masked one-hot scatter into per-chunk partials. Thread tid owns h-column
// tid exclusively -> race-free, no atomics, deterministic.
// ---------------------------------------------------------------------------
__global__ void __launch_bounds__(512)
scatter_partial(const float* __restrict__ s, const float* __restrict__ o,
                const int* __restrict__ index, const int* __restrict__ rel_lens,
                float* __restrict__ partial) {
  __shared__ float acc[OH];              // 36 KB
  const int tid   = threadIdx.x;         // == h
  const int b     = blockIdx.x;
  const int chunk = blockIdx.y;
  for (int i = tid; i < OH; i += 512) acc[i] = 0.f;
  __syncthreads();

  const int rl = rel_lens[b];
  const int t0 = chunk * (T_DIM / NCHUNK);
  for (int tt = 0; tt < T_DIM / NCHUNK; ++tt) {
    const int t = t0 + tt;
    if (t < rl) {
      const size_t row = (size_t)b * T_DIM + t;
      const int is = index[row * 2 + 0];
      const int io = index[row * 2 + 1];
      acc[is * H_DIM + tid] += s[row * H_DIM + tid];
      acc[io * H_DIM + tid] += o[row * H_DIM + tid];
    }
  }
  __syncthreads();
  float* dst = partial + ((size_t)b * NCHUNK + chunk) * OH;
  for (int i = tid; i < OH; i += 512) dst[i] = acc[i];
}

// ---------------------------------------------------------------------------
// Reduce chunks (fixed order) + cumulative sum over the batch axis.
// ---------------------------------------------------------------------------
__global__ void cumsum_pooled(const float* __restrict__ partial,
                              float* __restrict__ pooled) {
  const int idx = blockIdx.x * blockDim.x + threadIdx.x;   // 0..OH-1
  if (idx >= OH) return;
  float run = 0.f;
  for (int b = 0; b < B_DIM; ++b) {
    float c = 0.f;
#pragma unroll
    for (int ch = 0; ch < NCHUNK; ++ch)
      c += partial[((size_t)b * NCHUNK + ch) * OH + idx];
    run += c;
    pooled[(size_t)b * OH + idx] = run;
  }
}

// ---------------------------------------------------------------------------
// Host-side orchestration
// ---------------------------------------------------------------------------
extern "C" void kernel_launch(void* const* d_in, const int* in_sizes, int n_in,
                              void* d_out, int out_size, void* d_ws, size_t ws_size,
                              hipStream_t stream) {
  (void)in_sizes; (void)n_in; (void)out_size; (void)ws_size;

  const float* trip     = (const float*)d_in[0];
  const int*   index    = (const int*)  d_in[1];
  const int*   rel_lens = (const int*)  d_in[2];
  const float* w1a      = (const float*)d_in[3];
  const float* b1a      = (const float*)d_in[4];
  const float* w1b      = (const float*)d_in[5];
  const float* b1b      = (const float*)d_in[6];
  const float* w2a      = (const float*)d_in[7];
  const float* b2a      = (const float*)d_in[8];
  const float* w2b      = (const float*)d_in[9];
  const float* b2b      = (const float*)d_in[10];

  float* out     = (float*)d_out;
  float* new_obj = out;                                  // (32,18,512)
  float* p_out   = out + (size_t)B_DIM * O_DIM * DOUT;   // (32,2048,512)

  // ---- workspace carve-up ----
  char*  ws  = (char*)d_ws;
  size_t off = 0;
  auto take = [&](size_t bytes) -> char* {
    char* p = ws + off;
    off += (bytes + 255) & ~(size_t)255;
    return p;
  };
  const int K1p = 1504;  // 1500 padded to a multiple of 32
  float* h_buf = (float*)take((size_t)MT * H_DIM * 4);
  float* s_buf = (float*)take((size_t)MT * H_DIM * 4);
  float* o_buf = (float*)take((size_t)MT * H_DIM * 4);
  unsigned short* w1aTh = (unsigned short*)take((size_t)H_DIM * K1p * 2);
  unsigned short* w1aTl = (unsigned short*)take((size_t)H_DIM * K1p * 2);
  unsigned short* w1bTh = (unsigned short*)take((size_t)1536 * 512 * 2);
  unsigned short* w1bTl = (unsigned short*)take((size_t)1536 * 512 * 2);
  unsigned short* w2aTh = (unsigned short*)take((size_t)512 * 512 * 2);
  unsigned short* w2aTl = (unsigned short*)take((size_t)512 * 512 * 2);
  unsigned short* w2bTh = (unsigned short*)take((size_t)512 * 512 * 2);
  unsigned short* w2bTl = (unsigned short*)take((size_t)512 * 512 * 2);
  float* partial = (float*)take((size_t)B_DIM * NCHUNK * OH * 4);
  float* pooled  = (float*)take((size_t)B_DIM * OH * 4);
  float* h2_buf  = (float*)take((size_t)B_DIM * O_DIM * H_DIM * 4);

  const int BIG = 1 << 30;

  // ---- 1) transpose + hi/lo split of all weights ----
  { int tot = H_DIM * K1p;
    split_weight_T<<<(tot + 255) / 256, 256, 0, stream>>>(w1a, DIN1_K, H_DIM, K1p, w1aTh, w1aTl, tot); }
  { int tot = 1536 * 512;
    split_weight_T<<<(tot + 255) / 256, 256, 0, stream>>>(w1b, 512, 1536, 512, w1bTh, w1bTl, tot); }
  { int tot = 512 * 512;
    split_weight_T<<<(tot + 255) / 256, 256, 0, stream>>>(w2a, 512, 512, 512, w2aTh, w2aTl, tot);
    split_weight_T<<<(tot + 255) / 256, 256, 0, stream>>>(w2b, 512, 512, 512, w2bTh, w2bTl, tot); }

  // ---- 2) GEMM1: h = relu(trip @ w1a + b1a)   (65536 x 1500 x 512) ----
  gemm_bf16x3_relu<<<dim3(H_DIM / 256, MT / 64), 256, 0, stream>>>(
      trip, DIN1_K, DIN1_K, K1p, w1aTh, w1aTl, b1a,
      h_buf, h_buf, h_buf, BIG, BIG, H_DIM, H_DIM, H_DIM);

  // ---- 3) GEMM2: new_t = relu(h @ w1b + b1b), columns routed:
  //         [0,512) -> s_buf, [512,1024) -> p (d_out), [1024,1536) -> o_buf --
  gemm_bf16x3_relu<<<dim3(1536 / 256, MT / 64), 256, 0, stream>>>(
      h_buf, H_DIM, 512, 512, w1bTh, w1bTl, b1b,
      s_buf, p_out, o_buf, 512, 1024, H_DIM, H_DIM, H_DIM);

  // ---- 4) masked one-hot scatter into per-chunk partials ----
  scatter_partial<<<dim3(B_DIM, NCHUNK), 512, 0, stream>>>(
      s_buf, o_buf, index, rel_lens, partial);

  // ---- 5) chunk reduction + batch cumsum -> pooled (32,18,512) ----
  cumsum_pooled<<<(OH + 511) / 512, 512, 0, stream>>>(partial, pooled);

  // ---- 6) h2 = relu(pooled @ w2a + b2a)   (576 x 512 x 512) ----
  gemm_bf16x3_relu<<<dim3(512 / 256, (B_DIM * O_DIM) / 64), 256, 0, stream>>>(
      pooled, H_DIM, 512, 512, w2aTh, w2aTl, b2a,
      h2_buf, h2_buf, h2_buf, BIG, BIG, H_DIM, H_DIM, H_DIM);

  // ---- 7) new_obj = relu(h2 @ w2b + b2b) -> d_out[0 .. 294912) ----
  gemm_bf16x3_relu<<<dim3(512 / 256, (B_DIM * O_DIM) / 64), 256, 0, stream>>>(
      h2_buf, H_DIM, 512, 512, w2bTh, w2bTl, b2b,
      new_obj, new_obj, new_obj, BIG, BIG, DOUT, DOUT, DOUT);
}
